// UNetV2_5703716569163
// MI455X (gfx1250) — matrix operand order repaired and checked by reading, not compiled, not run on hardware
//
#include <hip/hip_runtime.h>

typedef __attribute__((ext_vector_type(16))) _Float16 v16h;
typedef __attribute__((ext_vector_type(8)))  _Float16 v8h;
typedef __attribute__((ext_vector_type(8)))  float    v8f;

#define NORM_EPS   0.001f
#define LEAK_SLOPE 0.01f

static __device__ inline v16h join_h16(v8h lo, v8h hi) {
  union { struct { v8h lo, hi; } s; v16h v; } u;
  u.s.lo = lo; u.s.hi = hi;
  return u.v;
}

// ---------------------------------------------------------------------------
// Utility / elementwise kernels
// ---------------------------------------------------------------------------
__global__ void zero_f32_kernel(float* __restrict__ p, int n) {
  int i = blockIdx.x * blockDim.x + threadIdx.x;
  if (i < n) p[i] = 0.0f;
}

__global__ void zero_f16_kernel(_Float16* __restrict__ p, int n) {
  int i = blockIdx.x * blockDim.x + threadIdx.x;
  if (i < n) p[i] = (_Float16)0.0f;
}

// f32 weights (3,3,3,cin,cout) -> f16 transposed [cout][Ktot_pad], K-major,
// K = tap*cin_pad + ci, zero padded in both ci (>=cin) and K (>=27*cin_pad).
__global__ void prep_w_kernel(const float* __restrict__ w, _Float16* __restrict__ o,
                              int cin, int cshift, int cout, int Ktotp) {
  int k  = blockIdx.x * blockDim.x + threadIdx.x;
  int co = blockIdx.y;
  if (k >= Ktotp) return;
  int tap = k >> cshift;
  int ci  = k & ((1 << cshift) - 1);
  float v = (tap < 27 && ci < cin) ? w[((size_t)tap * cin + ci) * cout + co] : 0.0f;
  o[(size_t)co * Ktotp + k] = (_Float16)v;
}

// Scatter sparse voxel features into dense (Z,Y,X,Cp) f16 grid + binary mask.
__global__ void scatter_kernel(const int* __restrict__ coords,
                               const float* __restrict__ feats,
                               _Float16* __restrict__ dense,
                               float* __restrict__ mask,
                               int nvox, int H, int W, int Cp, int Cin) {
  int i = blockIdx.x * blockDim.x + threadIdx.x;
  if (i >= nvox) return;
  int z = coords[i * 4 + 3];
  int y = coords[i * 4 + 2];
  int x = coords[i * 4 + 1];
  int lin = (z * H + y) * W + x;
  mask[lin] = 1.0f;
  for (int c = 0; c < Cin; ++c)
    dense[(size_t)lin * Cp + c] = (_Float16)feats[i * Cin + c];
}

// Gather dense act (f16, C=16) back to sparse voxel list (f32 out).
__global__ void gather_kernel(const int* __restrict__ coords,
                              const _Float16* __restrict__ act,
                              float* __restrict__ out,
                              int nvox, int H, int W) {
  int i = blockIdx.x * blockDim.x + threadIdx.x;
  if (i >= (nvox << 4)) return;
  int v = i >> 4, c = i & 15;
  int z = coords[v * 4 + 3];
  int y = coords[v * 4 + 2];
  int x = coords[v * 4 + 1];
  int lin = (z * H + y) * W + x;
  out[i] = (float)act[((size_t)lin << 4) + c];
}

// 3x3x3 stride-2 max-pool (pad 1) of binary mask.
__global__ void mask_down_kernel(const float* __restrict__ m, float* __restrict__ o,
                                 int iD, int iH, int iW) {
  int oD = iD >> 1, oH = iH >> 1, oW = iW >> 1;
  int n = oD * oH * oW;
  int i = blockIdx.x * blockDim.x + threadIdx.x;
  if (i >= n) return;
  int oz = i / (oH * oW);
  int r  = i - oz * oH * oW;
  int oy = r / oW;
  int ox = r - oy * oW;
  float v = 0.0f;
  for (int kz = 0; kz < 3; ++kz)
    for (int ky = 0; ky < 3; ++ky)
      for (int kx = 0; kx < 3; ++kx) {
        int iz = 2 * oz + kz - 1, iy = 2 * oy + ky - 1, ix = 2 * ox + kx - 1;
        if (iz >= 0 && iz < iD && iy >= 0 && iy < iH && ix >= 0 && ix < iW)
          v = fmaxf(v, m[(iz * iH + iy) * iW + ix]);
      }
  o[i] = v;
}

// Per-channel sum / sum-of-squares over the masked conv output.
// Flat coalesced sweep; per-channel LDS bins via ds_add_f32 shared atomics.
__global__ void stats_kernel(const float* __restrict__ y, float* __restrict__ sums,
                             int total, int C, int cshift) {
  __shared__ float ls[512];  // 2 * C_max(256)
  int tid = threadIdx.x;
  for (int j = tid; j < 2 * C; j += blockDim.x) ls[j] = 0.0f;
  __syncthreads();
  int cmask = C - 1;
  for (int i = blockIdx.x * blockDim.x + tid; i < total; i += gridDim.x * blockDim.x) {
    float t = y[i];
    int c = i & cmask;
    atomicAdd(&ls[c], t);
    atomicAdd(&ls[C + c], t * t);
  }
  __syncthreads();
  for (int j = tid; j < 2 * C; j += blockDim.x)
    atomicAdd(&sums[j], ls[j]);
}

__global__ void mask_count_kernel(const float* __restrict__ m, float* __restrict__ cnt,
                                  int nvox) {
  int tid = threadIdx.x;
  float a = 0.0f;
  for (int v = blockIdx.x * blockDim.x + tid; v < nvox; v += gridDim.x * blockDim.x)
    a += m[v];
  __shared__ float s1[256];
  s1[tid] = a;
  __syncthreads();
  for (int s = 128; s > 0; s >>= 1) {
    if (tid < s) s1[tid] += s1[tid + s];
    __syncthreads();
  }
  if (tid == 0) atomicAdd(cnt, s1[0]);
}

// Instance-norm (masked) + optional residual + leaky ReLU, f32 -> f16 act.
__global__ void norm_act_kernel(const float* __restrict__ y,
                                const float* __restrict__ sums,
                                const float* __restrict__ cnt,
                                const float* __restrict__ g,
                                const float* __restrict__ b,
                                const float* __restrict__ mask,
                                const _Float16* __restrict__ res,
                                _Float16* __restrict__ out,
                                int total, int C, int cshift, int hasRes) {
  int i = blockIdx.x * blockDim.x + threadIdx.x;
  if (i >= total) return;
  int v = i >> cshift, c = i & (C - 1);
  float n    = cnt[0];
  float mean = sums[c] / n;
  float var  = fmaxf(sums[C + c] / n - mean * mean, 0.0f);
  float val  = (y[i] - mean) * rsqrtf(var + NORM_EPS) * g[c] + b[c];
  val *= mask[v];
  if (hasRes) val += (float)res[i];
  out[i] = (_Float16)(val > 0.0f ? val : LEAK_SLOPE * val);
}

// cat = [a , b] along channels (C power of two; c2shift = log2(2C))
__global__ void concat_kernel(const _Float16* __restrict__ a, const _Float16* __restrict__ b,
                              _Float16* __restrict__ o, int total, int C, int c2shift) {
  int i = blockIdx.x * blockDim.x + threadIdx.x;
  if (i >= total) return;
  int v = i >> c2shift, c = i & ((1 << c2shift) - 1);
  o[i] = (c < C) ? a[((size_t)v << (c2shift - 1)) + c]
                 : b[((size_t)v << (c2shift - 1)) + (c - C)];
}

// out = xm + channel_reduce(cat): out[j] = xm[j] + cat[2j] + cat[2j+1]
__global__ void add_chred_kernel(const _Float16* __restrict__ xm,
                                 const _Float16* __restrict__ cat,
                                 _Float16* __restrict__ o, int total, int C, int cshift) {
  int i = blockIdx.x * blockDim.x + threadIdx.x;
  if (i >= total) return;
  int v = i >> cshift, c = i & (C - 1);
  size_t cb = ((size_t)v << (cshift + 1)) + 2 * c;
  float s = (float)xm[i] + (float)cat[cb] + (float)cat[cb + 1];
  o[i] = (_Float16)s;
}

// ---------------------------------------------------------------------------
// Implicit-GEMM 3D conv via V_WMMA_F32_16X16X32_F16.
// One wave per (32 output voxels) x (16 output channels): two 16x16 D tiles,
// B fragment reused across both WMMAs per K-chunk.
//  - Every level's voxel count is a multiple of 32 -> NO store guards needed.
//  - All dims / channel counts are powers of two -> shift/mask addressing.
//  - Each lane's 16 K-values per chunk sit inside ONE kernel tap, so its
//    input halves are contiguous: 2x b128 global load + 2x b128 LDS store.
//  - Weights are pre-transposed [cout][Ktot_pad]; the B fragment is 32
//    contiguous bytes per lane -> 2x b128 global load, no LDS.
//  - A fragments use contiguous k order (k = 16*half + s, as B does), which is
//    one contiguous 16-half run per lane -> 2x b128 LDS load each.
//   MODE 0: stride-1 conv, MODE 1: stride-2 down, MODE 2: stride-2 transposed.
// ---------------------------------------------------------------------------
template <int MODE>
__launch_bounds__(32)
__global__ void conv3d_wmma_kernel(const _Float16* __restrict__ x,
                                   const _Float16* __restrict__ wT,
                                   const float* __restrict__ mask,
                                   float* __restrict__ y,
                                   int log2W, int log2HW,
                                   int iD, int iH, int iW,
                                   int log2iW, int log2iHW,
                                   int cshift, int Cout, int Ktotp) {
  __shared__ _Float16 Ash[2][16 * 32];

  const int lane    = threadIdx.x;     // 0..31
  const int half16  = lane >> 4;       // 0/1
  const int mrow    = lane & 15;
  const int voxBase = blockIdx.x << 5; // 32 output voxels per wave (exact)
  const int nbase   = blockIdx.y << 4;
  const int Cin     = 1 << cshift;
  const int Ktot    = 27 << cshift;

  // A staging: lane covers row (lane>>1) of each tile, K-halves (lane&1)*16..+16
  const int lrow   = lane >> 1;
  const int kkBase = (lane & 1) << 4;

  int ozv[2], oyv[2], oxv[2];
#pragma unroll
  for (int t = 0; t < 2; ++t) {
    int vox = voxBase + t * 16 + lrow;
    ozv[t] = vox >> log2HW;
    oyv[t] = (vox >> log2W) & ((1 << (log2HW - log2W)) - 1);
    oxv[t] = vox & ((1 << log2W) - 1);
  }

  const _Float16* wrow = wT + (size_t)(nbase + mrow) * Ktotp + (half16 << 4);

  v8f acc0 = {}, acc1 = {};
  const v8h zro = {};

  for (int k0 = 0; k0 < Ktot; k0 += 32) {
    // ---- stage A tiles: one tap per lane per chunk, contiguous channels ----
    const int kbl = k0 + kkBase;
    const int tap = kbl >> cshift;
    const int ci0 = kbl & (Cin - 1);
    const int kz  = tap / 9;
    const int t9  = tap - kz * 9;
    const int ky  = t9 / 3;
    const int kx  = t9 - ky * 3;
#pragma unroll
    for (int t = 0; t < 2; ++t) {
      int iz, iy, ix;
      int ok = (tap < 27);
      if (MODE == 0) {
        iz = ozv[t] + kz - 1; iy = oyv[t] + ky - 1; ix = oxv[t] + kx - 1;
      } else if (MODE == 1) {
        iz = 2 * ozv[t] + kz - 1; iy = 2 * oyv[t] + ky - 1; ix = 2 * oxv[t] + kx - 1;
      } else {
        int tz = ozv[t] + kz - 1, ty = oyv[t] + ky - 1, tx = oxv[t] + kx - 1;
        int o3 = tz | ty | tx;
        ok &= (o3 >= 0) & ((o3 & 1) == 0);
        iz = tz >> 1; iy = ty >> 1; ix = tx >> 1;
      }
      ok &= ((unsigned)iz < (unsigned)iD) & ((unsigned)iy < (unsigned)iH) &
            ((unsigned)ix < (unsigned)iW);
      int av = (iz << log2iHW) + (iy << log2iW) + ix;
      av = ok ? av : 0;
      const _Float16* src = x + (((size_t)av << cshift) + ci0);
      v8h d0 = *(const v8h*)(src);
      v8h d1 = *(const v8h*)(src + 8);
      if (!ok) { d0 = zro; d1 = zro; }
      *(v8h*)(&Ash[t][lrow * 32 + kkBase])     = d0;
      *(v8h*)(&Ash[t][lrow * 32 + kkBase + 8]) = d1;
    }
    __syncthreads();

    // ---- B fragment straight from transposed weights (K-contiguous) ----
    const _Float16* bsrc = wrow + k0;
    v16h bf = join_h16(*(const v8h*)(bsrc), *(const v8h*)(bsrc + 8));

    // ---- A fragments: one contiguous 16-half run per lane (k = 16*half + s) ----
    v16h af0 = join_h16(*(const v8h*)(&Ash[0][mrow * 32 + half16 * 16]),
                        *(const v8h*)(&Ash[0][mrow * 32 + half16 * 16 + 8]));
    v16h af1 = join_h16(*(const v8h*)(&Ash[1][mrow * 32 + half16 * 16]),
                        *(const v8h*)(&Ash[1][mrow * 32 + half16 * 16 + 8]));

    acc0 = __builtin_amdgcn_wmma_f32_16x16x32_f16(false, af0, false, bf,
                                                  (short)0, acc0, false, false);
    acc1 = __builtin_amdgcn_wmma_f32_16x16x32_f16(false, af1, false, bf,
                                                  (short)0, acc1, false, false);
    __syncthreads();
  }

  // ---- epilogue: unconditional masked stores (tiles are always full) ----
  // mask values for this lane's 8 rows are contiguous: vector loads.
  float mv0[8], mv1[8];
  {
    const float* mp0 = mask + voxBase + half16 * 8;
    ((float4*)mv0)[0] = *(const float4*)(mp0);
    ((float4*)mv0)[1] = *(const float4*)(mp0 + 4);
    const float* mp1 = mp0 + 16;
    ((float4*)mv1)[0] = *(const float4*)(mp1);
    ((float4*)mv1)[1] = *(const float4*)(mp1 + 4);
  }
  float* yrow = y + (size_t)(voxBase + half16 * 8) * Cout + nbase + mrow;
#pragma unroll
  for (int r = 0; r < 8; ++r) {
    yrow[(size_t)r * Cout]        = acc0[r] * mv0[r];
    yrow[(size_t)(16 + r) * Cout] = acc1[r] * mv1[r];
  }
}

// ---------------------------------------------------------------------------
// Host-side orchestration of the full U-Net forward pass.
// ---------------------------------------------------------------------------
extern "C" void kernel_launch(void* const* d_in, const int* in_sizes, int n_in,
                              void* d_out, int out_size, void* d_ws, size_t ws_size,
                              hipStream_t stream) {
  (void)n_in; (void)out_size; (void)ws_size;

  const int*   coords = (const int*)d_in[0];
  const float* feats  = (const float*)d_in[1];
  int pi = 3;  // param leaf cursor (skips batch_size at d_in[2])
  const int NV = in_sizes[0] / 4;

  // Bump allocator over workspace.
  char* base = (char*)d_ws;
  size_t off = 0;
  auto alloc = [&](size_t bytes) -> void* {
    void* p = base + off;
    off += (bytes + 255) & ~(size_t)255;
    return p;
  };
  auto lg = [](int v) { int s = 0; while ((1 << s) < v) ++s; return s; };

  struct Lvl { int D, H, W, nv; };
  Lvl L1{32, 64, 64, 32 * 64 * 64};
  Lvl L2{16, 32, 32, 16 * 32 * 32};
  Lvl L3{8, 16, 16, 8 * 16 * 16};
  Lvl L4{4, 8, 8, 4 * 8 * 8};
  Lvl L5{2, 4, 4, 2 * 4 * 4};

  // Shared scratch (reused across the whole pass; stream is in-order).
  float* f32tmp  = (float*)alloc((size_t)L1.nv * 16 * sizeof(float));
  float* statbuf = (float*)alloc((2 * 256 + 1) * sizeof(float));

  struct PabP { const float* g; const float* b; _Float16* w; int cshift; int cout; int Ktotp; };
  struct BbP  { PabP p1, p2; };

  auto get_pab = [&](int cin, int cout) -> PabP {
    const float* w = (const float*)d_in[pi++];
    const float* g = (const float*)d_in[pi++];
    const float* b = (const float*)d_in[pi++];
    int cinp   = (cin + 15) & ~15;
    int cshift = lg(cinp);
    int Ktotp  = ((27 * cinp) + 31) & ~31;
    _Float16* wh = (_Float16*)alloc((size_t)cout * Ktotp * sizeof(_Float16));
    prep_w_kernel<<<dim3((Ktotp + 255) / 256, cout), 256, 0, stream>>>(
        w, wh, cin, cshift, cout, Ktotp);
    PabP r; r.g = g; r.b = b; r.w = wh; r.cshift = cshift; r.cout = cout; r.Ktotp = Ktotp;
    return r;
  };
  auto get_bb = [&](int c) -> BbP {
    BbP r;
    r.p1 = get_pab(c, c);  // w1,g1,b1
    r.p2 = get_pab(c, c);  // w2,g2,b2
    return r;
  };

  auto aact = [&](Lvl l, int C) -> _Float16* {
    return (_Float16*)alloc((size_t)l.nv * C * sizeof(_Float16));
  };

  auto norm_pipeline = [&](const float* ytmp, const float* g, const float* b,
                           const float* mask, const _Float16* res,
                           _Float16* out, int nv, int C) {
    int cs = lg(C);
    zero_f32_kernel<<<(2 * C + 1 + 255) / 256, 256, 0, stream>>>(statbuf, 2 * C + 1);
    int mb = (nv + 255) / 256; if (mb > 64) mb = 64; if (mb < 1) mb = 1;
    mask_count_kernel<<<mb, 256, 0, stream>>>(mask, statbuf + 2 * C, nv);
    int tot = nv * C;
    int sb = (tot + 255) / 256; if (sb > 256) sb = 256; if (sb < 1) sb = 1;
    stats_kernel<<<sb, 256, 0, stream>>>(ytmp, statbuf, tot, C, cs);
    norm_act_kernel<<<(tot + 255) / 256, 256, 0, stream>>>(
        ytmp, statbuf, statbuf + 2 * C, g, b, mask, res, out, tot, C, cs, res ? 1 : 0);
  };

  auto pab_op = [&](const _Float16* xin, const PabP& p, const float* mask,
                    Lvl lo, Lvl li, int mode, const _Float16* res, _Float16* out) {
    dim3 grid(lo.nv / 32, p.cout / 16);
    int l2W  = lg(lo.W), l2HW = lg(lo.H) + lg(lo.W);
    int l2iW = lg(li.W), l2iHW = lg(li.H) + lg(li.W);
    if (mode == 0)
      conv3d_wmma_kernel<0><<<grid, 32, 0, stream>>>(xin, p.w, mask, f32tmp,
          l2W, l2HW, li.D, li.H, li.W, l2iW, l2iHW, p.cshift, p.cout, p.Ktotp);
    else if (mode == 1)
      conv3d_wmma_kernel<1><<<grid, 32, 0, stream>>>(xin, p.w, mask, f32tmp,
          l2W, l2HW, li.D, li.H, li.W, l2iW, l2iHW, p.cshift, p.cout, p.Ktotp);
    else
      conv3d_wmma_kernel<2><<<grid, 32, 0, stream>>>(xin, p.w, mask, f32tmp,
          l2W, l2HW, li.D, li.H, li.W, l2iW, l2iHW, p.cshift, p.cout, p.Ktotp);
    norm_pipeline(f32tmp, p.g, p.b, mask, res, out, lo.nv, p.cout);
  };

  auto bb_op = [&](const _Float16* xin, const BbP& p, const float* mask,
                   Lvl l, _Float16* out) {
    _Float16* t = aact(l, p.p1.cout);
    pab_op(xin, p.p1, mask, l, l, 0, nullptr, t);
    pab_op(t, p.p2, mask, l, l, 0, xin, out);  // residual + lrelu
  };

  // -------------------- input scatter --------------------
  _Float16* dense = aact(L1, 16);
  float* mask1 = (float*)alloc((size_t)L1.nv * sizeof(float));
  zero_f16_kernel<<<(L1.nv * 16 + 255) / 256, 256, 0, stream>>>(dense, L1.nv * 16);
  zero_f32_kernel<<<(L1.nv + 255) / 256, 256, 0, stream>>>(mask1, L1.nv);
  scatter_kernel<<<(NV + 255) / 256, 256, 0, stream>>>(coords, feats, dense, mask1,
                                                       NV, 64, 64, 16, 5);

  // -------------------- encoder --------------------
  PabP p_in = get_pab(5, 16);
  _Float16* h1 = aact(L1, 16);
  pab_op(dense, p_in, mask1, L1, L1, 0, nullptr, h1);

  BbP pb11 = get_bb(16), pb12 = get_bb(16);
  _Float16* t1 = aact(L1, 16);
  bb_op(h1, pb11, mask1, L1, t1);
  _Float16* x1 = aact(L1, 16);
  bb_op(t1, pb12, mask1, L1, x1);

  auto down_op = [&](const _Float16* xin, const PabP& p, const float* mIn,
                     Lvl lin, Lvl lout, float* mOut, _Float16* out) {
    mask_down_kernel<<<(lout.nv + 255) / 256, 256, 0, stream>>>(mIn, mOut,
                                                                lin.D, lin.H, lin.W);
    pab_op(xin, p, mOut, lout, lin, 1, nullptr, out);
  };

  PabP p2d = get_pab(16, 32);
  float* mask2 = (float*)alloc((size_t)L2.nv * sizeof(float));
  _Float16* h2 = aact(L2, 32);
  down_op(x1, p2d, mask1, L1, L2, mask2, h2);
  BbP pb21 = get_bb(32), pb22 = get_bb(32);
  _Float16* t2 = aact(L2, 32);
  bb_op(h2, pb21, mask2, L2, t2);
  _Float16* x2 = aact(L2, 32);
  bb_op(t2, pb22, mask2, L2, x2);

  PabP p3d = get_pab(32, 64);
  float* mask3 = (float*)alloc((size_t)L3.nv * sizeof(float));
  _Float16* h3 = aact(L3, 64);
  down_op(x2, p3d, mask2, L2, L3, mask3, h3);
  BbP pb31 = get_bb(64), pb32 = get_bb(64);
  _Float16* t3 = aact(L3, 64);
  bb_op(h3, pb31, mask3, L3, t3);
  _Float16* x3 = aact(L3, 64);
  bb_op(t3, pb32, mask3, L3, x3);

  PabP p4d = get_pab(64, 64);
  float* mask4 = (float*)alloc((size_t)L4.nv * sizeof(float));
  _Float16* h4 = aact(L4, 64);
  down_op(x3, p4d, mask3, L3, L4, mask4, h4);
  BbP pb41 = get_bb(64), pb42 = get_bb(64);
  _Float16* t4 = aact(L4, 64);
  bb_op(h4, pb41, mask4, L4, t4);
  _Float16* x4 = aact(L4, 64);
  bb_op(t4, pb42, mask4, L4, x4);

  PabP p5d = get_pab(64, 128);
  float* mask5 = (float*)alloc((size_t)L5.nv * sizeof(float));
  _Float16* h5 = aact(L5, 128);
  down_op(x4, p5d, mask4, L4, L5, mask5, h5);
  BbP pb51 = get_bb(128), pb52 = get_bb(128);
  _Float16* t5 = aact(L5, 128);
  bb_op(h5, pb51, mask5, L5, t5);
  _Float16* x5 = aact(L5, 128);
  bb_op(t5, pb52, mask5, L5, x5);

  // -------------------- decoder --------------------
  auto ur_op = [&](const _Float16* lateral, const _Float16* bottom, const float* mask,
                   Lvl l, int c, float* upMask, Lvl lup, int cInv) -> _Float16* {
    BbP pt = get_bb(c);
    PabP pm = get_pab(2 * c, c);
    _Float16* xtr = aact(l, c);
    bb_op(lateral, pt, mask, l, xtr);
    _Float16* cat = aact(l, 2 * c);
    int tot = l.nv * 2 * c;
    concat_kernel<<<(tot + 255) / 256, 256, 0, stream>>>(bottom, xtr, cat, tot, c,
                                                         lg(2 * c));
    _Float16* xm = aact(l, c);
    pab_op(cat, pm, mask, l, l, 0, nullptr, xm);
    _Float16* xs = aact(l, c);
    int tot2 = l.nv * c;
    add_chred_kernel<<<(tot2 + 255) / 256, 256, 0, stream>>>(xm, cat, xs, tot2, c, lg(c));
    PabP pinv = get_pab(c, cInv);
    _Float16* up = aact(lup, cInv);
    pab_op(xs, pinv, upMask, lup, l, 2, nullptr, up);  // transposed conv
    return up;
  };

  _Float16* up5 = ur_op(x5, x5, mask5, L5, 128, mask4, L4, 64);
  _Float16* up4 = ur_op(x4, up5, mask4, L4, 64, mask3, L3, 64);
  _Float16* up3 = ur_op(x3, up4, mask3, L3, 64, mask2, L2, 32);
  _Float16* up2 = ur_op(x2, up3, mask2, L2, 32, mask1, L1, 16);

  // final stage (bb instead of transposed conv)
  BbP pt1 = get_bb(16);
  PabP pm1 = get_pab(32, 16);
  _Float16* xtr1 = aact(L1, 16);
  bb_op(x1, pt1, mask1, L1, xtr1);
  _Float16* cat1 = aact(L1, 32);
  {
    int tot = L1.nv * 32;
    concat_kernel<<<(tot + 255) / 256, 256, 0, stream>>>(up2, xtr1, cat1, tot, 16, 5);
  }
  _Float16* xm1 = aact(L1, 16);
  pab_op(cat1, pm1, mask1, L1, L1, 0, nullptr, xm1);
  _Float16* xs1 = aact(L1, 16);
  {
    int tot = L1.nv * 16;
    add_chred_kernel<<<(tot + 255) / 256, 256, 0, stream>>>(xm1, cat1, xs1, tot, 16, 4);
  }
  BbP pfin = get_bb(16);
  _Float16* up1 = aact(L1, 16);
  bb_op(xs1, pfin, mask1, L1, up1);

  // -------------------- gather sparse output --------------------
  gather_kernel<<<(NV * 16 + 255) / 256, 256, 0, stream>>>(coords, up1, (float*)d_out,
                                                           NV, 64, 64);
}
